// vector_quantizer_56513179681061
// MI455X (gfx1250) — matrix operand order repaired and checked by reading, not compiled, not run on hardware
//
#include <hip/hip_runtime.h>

// ---------------------------------------------------------------------------
// VQ-VAE vector quantizer for MI455X (gfx1250, wave32, WMMA)
// inputs:  (16, 64, 4096) f32 ; codebook (1024, 64) f32
// ---------------------------------------------------------------------------

typedef __attribute__((ext_vector_type(16))) __bf16 v16bf;
typedef __attribute__((ext_vector_type(8)))  float  v8f;

#define NUM_EMB   1024
#define DIM       64
#define NBATCH    16
#define LPERB     4096
#define NROWS     (NBATCH * LPERB)     // 65536
#define TILE_M    128
#define CHUNK_K   256                  // codes resident in LDS per chunk
#define XPAD      72                   // padded LDS row (bf16 elems) to spread banks
#define WPAD      72

#define DECAY     0.9f
#define COMMIT    0.1f

// ---- d_out layout (flat concat of the reference tuple, all f32) ----
#define OUT_OFF   0                    // (16,64,4096) = 4194304
#define LOSS_OFF  4194304
#define PERP_OFF  4194305
#define NEWW_OFF  4194306              // (1024,64) = 65536
#define ENC_OFF   4259842              // (65536,1)

// ---- workspace layout (bytes) ----
#define WS_WNORM  0                    // 1024 f32
#define WS_COUNTS 4096                 // 1024 f32
#define WS_KL     8192                 // 1 f32 (padded)
#define WS_DW     16384                // 1024*64 f32 = 256 KB
#define WS_WHI    278528               // 1024*64 bf16 = 128 KB
#define WS_WLO    409600               // 1024*64 bf16 = 128 KB
#define WS_IDX    538624               // 65536 i32 = 256 KB

// ===========================================================================
// Kernel 1: codebook prep — bf16 hi/lo split, ||w||^2, zero accumulators
// ===========================================================================
__global__ __launch_bounds__(DIM) void vq_prep(
    const float* __restrict__ W, float* __restrict__ wnorm,
    float* __restrict__ counts, float* __restrict__ dwbuf,
    float* __restrict__ klacc, __bf16* __restrict__ whi,
    __bf16* __restrict__ wlo)
{
    const int k = blockIdx.x;
    const int d = threadIdx.x;
    const float w = W[k * DIM + d];
    const __bf16 h = (__bf16)w;
    whi[k * DIM + d] = h;
    wlo[k * DIM + d] = (__bf16)(w - (float)h);
    dwbuf[k * DIM + d] = 0.0f;

    __shared__ float red[DIM];
    red[d] = w * w;
    __syncthreads();
    if (d == 0) {
        float s = 0.0f;
        #pragma unroll
        for (int i = 0; i < DIM; ++i) s += red[i];
        wnorm[k] = s;
        counts[k] = 0.0f;
        if (k == 0) klacc[0] = 0.0f;
    }
}

// ===========================================================================
// Kernel 2: distance GEMM (split-bf16 WMMA) + fused argmin + counts/dw
//   scores S = x_hi*w_hi + x_hi*w_lo + x_lo*w_hi  (~f32 accurate)
//   dist    = ||w||^2 - 2*S   (row-constant ||x||^2 dropped for argmin)
// ===========================================================================
__global__ __launch_bounds__(256) void vq_assign(
    const float* __restrict__ in_, const __bf16* __restrict__ whi,
    const __bf16* __restrict__ wlo, const float* __restrict__ wnorm,
    float* __restrict__ counts, float* __restrict__ dwbuf,
    int* __restrict__ idxout, float* __restrict__ encout)
{
    __shared__ __bf16 sXhi[TILE_M][XPAD];
    __shared__ __bf16 sXlo[TILE_M][XPAD];
    __shared__ __bf16 sWhi[CHUNK_K][WPAD];
    __shared__ __bf16 sWlo[CHUNK_K][WPAD];
    __shared__ float  sWn[CHUNK_K];
    __shared__ int    sRowIdx[TILE_M];

    const int t    = threadIdx.x;
    const int tile = blockIdx.x;                    // 512 tiles of 128 rows
    const int n0   = tile * TILE_M;
    const int b    = n0 / LPERB;                    // tiles never cross batch
    const int l0   = n0 % LPERB;
    const long inBase = (long)b * DIM * LPERB + l0; // + d*LPERB + m

    // ---- stage x tile (coalesced along m == l), split into bf16 hi/lo ----
    #pragma unroll 4
    for (int i = 0; i < 32; ++i) {
        int gi = t + i * 256;                       // 8192 elems
        int m = gi & 127, d = gi >> 7;
        float xv = in_[inBase + (long)d * LPERB + m];
        __bf16 h = (__bf16)xv;
        sXhi[m][d] = h;
        sXlo[m][d] = (__bf16)(xv - (float)h);
    }

    // ---- stage first codebook chunk ----
    {
        const __bf16* gh = whi;
        const __bf16* gl = wlo;
        #pragma unroll 4
        for (int i = 0; i < 64; ++i) {
            int gi = t + i * 256;                   // 16384 elems
            int r = gi >> 6, d = gi & 63;
            sWhi[r][d] = gh[gi];
            sWlo[r][d] = gl[gi];
        }
        sWn[t] = wnorm[t];
    }
    __syncthreads();

    const int lane = t & 31;
    const int wave = t >> 5;
    const int half = lane >> 4;                     // ISA 16-lane half
    const int nl   = lane & 15;
    const int m0   = wave * 16;                     // 16 rows per wave
    const int mrow = m0 + nl;                       // A: lanes L and L+16 share row

    // ---- A fragments (fixed for whole kernel), 16-bit A in contiguous k order (not the ISA table order):
    //      element e -> K = kbase + 16*half + e (contiguous, matches B)
    v16bf aHi[2], aLo[2];
    #pragma unroll
    for (int ks = 0; ks < 2; ++ks) {
        const int kb = ks * 32;
        const __bf16* p0h = &sXhi[mrow][kb + 16 * half];
        const __bf16* p1h = &sXhi[mrow][kb + 16 * half + 8];
        const __bf16* p0l = &sXlo[mrow][kb + 16 * half];
        const __bf16* p1l = &sXlo[mrow][kb + 16 * half + 8];
        #pragma unroll
        for (int e = 0; e < 8; ++e) {
            aHi[ks][e] = p0h[e];  aHi[ks][e + 8] = p1h[e];
            aLo[ks][e] = p0l[e];  aLo[ks][e + 8] = p1l[e];
        }
    }

    float best[8];
    int   bidx[8];
    #pragma unroll
    for (int r = 0; r < 8; ++r) { best[r] = 3.4e38f; bidx[r] = 0; }

    for (int chunk = 0; chunk < NUM_EMB / CHUNK_K; ++chunk) {
        if (chunk) {
            __syncthreads();
            const __bf16* gh = whi + chunk * CHUNK_K * DIM;
            const __bf16* gl = wlo + chunk * CHUNK_K * DIM;
            #pragma unroll 4
            for (int i = 0; i < 64; ++i) {
                int gi = t + i * 256;
                int r = gi >> 6, d = gi & 63;
                sWhi[r][d] = gh[gi];
                sWlo[r][d] = gl[gi];
            }
            sWn[t] = wnorm[chunk * CHUNK_K + t];
            __syncthreads();
        }

        for (int nt = 0; nt < CHUNK_K / 16; ++nt) {
            const int nrow = nt * 16 + nl;          // B: lane holds code row n
            // B fragments: element e -> d = ks*32 + 16*half + e (contiguous)
            v16bf bHi[2], bLo[2];
            #pragma unroll
            for (int ks = 0; ks < 2; ++ks) {
                const __bf16* qh = &sWhi[nrow][ks * 32 + 16 * half];
                const __bf16* ql = &sWlo[nrow][ks * 32 + 16 * half];
                #pragma unroll
                for (int e = 0; e < 16; ++e) { bHi[ks][e] = qh[e]; bLo[ks][e] = ql[e]; }
            }
            v8f c = {0.f, 0.f, 0.f, 0.f, 0.f, 0.f, 0.f, 0.f};
            c = __builtin_amdgcn_wmma_f32_16x16x32_bf16(false, aHi[0], false, bHi[0], (short)0, c, false, false);
            c = __builtin_amdgcn_wmma_f32_16x16x32_bf16(false, aHi[1], false, bHi[1], (short)0, c, false, false);
            c = __builtin_amdgcn_wmma_f32_16x16x32_bf16(false, aHi[0], false, bLo[0], (short)0, c, false, false);
            c = __builtin_amdgcn_wmma_f32_16x16x32_bf16(false, aHi[1], false, bLo[1], (short)0, c, false, false);
            c = __builtin_amdgcn_wmma_f32_16x16x32_bf16(false, aLo[0], false, bHi[0], (short)0, c, false, false);
            c = __builtin_amdgcn_wmma_f32_16x16x32_bf16(false, aLo[1], false, bHi[1], (short)0, c, false, false);

            const float wn  = sWn[nt * 16 + nl];
            const int kglob = chunk * CHUNK_K + nt * 16 + nl;
            // C layout: c[r] = S[m0 + r + 8*half][n-lane]
            #pragma unroll
            for (int r = 0; r < 8; ++r) {
                float dist = wn - 2.0f * c[r];
                if (dist < best[r]) { best[r] = dist; bidx[r] = kglob; }
            }
        }
    }

    // ---- argmin across the 16 N-lanes (xor masks stay inside each half) ----
    #pragma unroll
    for (int mask = 1; mask < 16; mask <<= 1) {
        #pragma unroll
        for (int r = 0; r < 8; ++r) {
            float ob = __shfl_xor(best[r], mask, 32);
            int   oi = __shfl_xor(bidx[r], mask, 32);
            if (ob < best[r] || (ob == best[r] && oi < bidx[r])) {
                best[r] = ob; bidx[r] = oi;
            }
        }
    }
    if (nl == 0) {
        #pragma unroll
        for (int r = 0; r < 8; ++r) {
            const int row = m0 + 8 * half + r;
            const int kf  = bidx[r];
            sRowIdx[row]  = kf;
            const int gn  = n0 + row;
            idxout[gn] = kf;
            encout[gn] = (float)kf;
            atomicAdd(&counts[kf], 1.0f);
        }
    }
    __syncthreads();

    // ---- dw[k][d] += x[n][d]  (coalesced re-read, scatter via L2 atomics) ----
    for (int rep = 0; rep < 32; ++rep) {
        const int m = t & 127;
        const int d = (t >> 7) + rep * 2;
        const int kf = sRowIdx[m];
        const float xv = in_[inBase + (long)d * LPERB + m];
        atomicAdd(&dwbuf[kf * DIM + d], xv);
    }
}

// ===========================================================================
// Kernel 3: KL loss + quantized (straight-through == quantized) output,
//           transposed back to (B, D, L) with coalesced stores
// ===========================================================================
__global__ __launch_bounds__(256) void vq_klout(
    const float* __restrict__ in_, const float* __restrict__ W,
    const int* __restrict__ idxws, float* __restrict__ out,
    float* __restrict__ klacc)
{
    __shared__ float sx[DIM][65];
    __shared__ float sq[DIM][65];
    __shared__ float skl;

    const int t  = threadIdx.x;
    const int bi = blockIdx.x;                      // 16 * 64 blocks
    const int b  = bi >> 6;
    const int l0 = (bi & 63) * 64;
    const long base = (long)b * DIM * LPERB + l0;

    #pragma unroll 4
    for (int it = 0; it < 16; ++it) {
        int d = it * 4 + (t >> 6);
        int l = t & 63;
        sx[d][l] = in_[base + (long)d * LPERB + l];
    }
    if (t == 0) skl = 0.0f;
    __syncthreads();

    const int row = t >> 2;                         // l within tile
    const int sub = t & 3;                          // 4 lanes per row
    const int n   = b * LPERB + l0 + row;
    const int k   = idxws[n];

    float xv[16], qv[16];
    #pragma unroll
    for (int i = 0; i < 16; ++i) {
        int d = sub * 16 + i;
        xv[i] = sx[d][row];
        float q = W[k * DIM + d];
        qv[i] = q;
        sq[d][row] = q;
    }

    float xm = -3.4e38f, qm = -3.4e38f;
    #pragma unroll
    for (int i = 0; i < 16; ++i) { xm = fmaxf(xm, xv[i]); qm = fmaxf(qm, qv[i]); }
    #pragma unroll
    for (int mask = 1; mask < 4; mask <<= 1) {
        xm = fmaxf(xm, __shfl_xor(xm, mask, 32));
        qm = fmaxf(qm, __shfl_xor(qm, mask, 32));
    }
    float sxs = 0.0f, sqs = 0.0f;
    #pragma unroll
    for (int i = 0; i < 16; ++i) { sxs += expf(xv[i] - xm); sqs += expf(qv[i] - qm); }
    #pragma unroll
    for (int mask = 1; mask < 4; mask <<= 1) {
        sxs += __shfl_xor(sxs, mask, 32);
        sqs += __shfl_xor(sqs, mask, 32);
    }
    const float lse = logf(sxs);
    float kl = 0.0f;
    #pragma unroll
    for (int i = 0; i < 16; ++i) {
        float e   = expf(xv[i] - xm);
        float pt  = e / sxs;
        float lpt = (xv[i] - xm) - lse;
        float pin = expf(qv[i] - qm) / sqs;
        kl += pt * (lpt - pin);                     // ref: p_tgt*(log p_tgt - p_in)
    }
    #pragma unroll
    for (int mask = 1; mask < 4; mask <<= 1) kl += __shfl_xor(kl, mask, 32);
    if (sub == 0) atomicAdd(&skl, kl);
    __syncthreads();

    #pragma unroll 4
    for (int it = 0; it < 16; ++it) {
        int d = it * 4 + (t >> 6);
        int l = t & 63;
        out[base + (long)d * LPERB + l] = sq[d][l];
    }
    if (t == 0) atomicAdd(klacc, skl);
}

// ===========================================================================
// Kernel 4: EMA / Laplace-smoothed cluster sizes, new codebook, loss, perp
// ===========================================================================
__global__ __launch_bounds__(NUM_EMB) void vq_finalize(
    const float* __restrict__ ecs, const float* __restrict__ emw,
    const float* __restrict__ counts, const float* __restrict__ dwbuf,
    const float* __restrict__ klacc, float* __restrict__ out)
{
    __shared__ float red[NUM_EMB];
    const int k = threadIdx.x;
    const float cnt = counts[k];
    const float cs  = ecs[k] * DECAY + (1.0f - DECAY) * cnt;

    red[k] = cs;
    __syncthreads();
    for (int s = NUM_EMB / 2; s > 0; s >>= 1) {
        if (k < s) red[k] += red[k + s];
        __syncthreads();
    }
    const float n = red[0];
    __syncthreads();

    const float avg = cnt / (float)NROWS;
    red[k] = avg * logf(avg + 1e-10f);
    __syncthreads();
    for (int s = NUM_EMB / 2; s > 0; s >>= 1) {
        if (k < s) red[k] += red[k + s];
        __syncthreads();
    }
    if (k == 0) {
        out[PERP_OFF] = expf(-red[0]);
        out[LOSS_OFF] = COMMIT * klacc[0] / (float)NBATCH;
    }

    const float csn = (cs + 1e-5f) / (n + (float)NUM_EMB * 1e-5f) * n;
    #pragma unroll 8
    for (int d = 0; d < DIM; ++d) {
        out[NEWW_OFF + k * DIM + d] =
            (emw[k * DIM + d] * DECAY + (1.0f - DECAY) * dwbuf[k * DIM + d]) / csn;
    }
}

// ===========================================================================
extern "C" void kernel_launch(void* const* d_in, const int* in_sizes, int n_in,
                              void* d_out, int out_size, void* d_ws, size_t ws_size,
                              hipStream_t stream)
{
    const float* in_ = (const float*)d_in[0];   // (16,64,4096)
    const float* W   = (const float*)d_in[1];   // (1024,64)
    const float* ecs = (const float*)d_in[2];   // (1024,)
    const float* emw = (const float*)d_in[3];   // (1024,64)
    float* out = (float*)d_out;

    char* ws = (char*)d_ws;
    float*  wnorm  = (float*)(ws + WS_WNORM);
    float*  counts = (float*)(ws + WS_COUNTS);
    float*  klacc  = (float*)(ws + WS_KL);
    float*  dwbuf  = (float*)(ws + WS_DW);
    __bf16* whi    = (__bf16*)(ws + WS_WHI);
    __bf16* wlo    = (__bf16*)(ws + WS_WLO);
    int*    idxws  = (int*)(ws + WS_IDX);

    vq_prep<<<dim3(NUM_EMB), dim3(DIM), 0, stream>>>(
        W, wnorm, counts, dwbuf, klacc, whi, wlo);

    vq_assign<<<dim3(NROWS / TILE_M), dim3(256), 0, stream>>>(
        in_, whi, wlo, wnorm, counts, dwbuf, idxws, out + ENC_OFF);

    vq_klout<<<dim3(NBATCH * (LPERB / 64)), dim3(256), 0, stream>>>(
        in_, W, idxws, out + OUT_OFF, klacc);

    vq_finalize<<<dim3(1), dim3(NUM_EMB), 0, stream>>>(
        ecs, emw, counts, dwbuf, klacc, out);
}
